// QKVAttention_11355893530795
// MI455X (gfx1250) — matrix operand order repaired and checked by reading, not compiled, not run on hardware
//
#include <hip/hip_runtime.h>
#include <math.h>

// ---------------------------------------------------------------------------
// Flash attention for qkv[64, 3*64, 2048] fp32 on gfx1250 (MI455X).
//   scores = (Q^T K)/sqrt(C), softmax over S, out[c,t] = sum_s P[t,s] V[c,s]
//
// Both GEMMs are computed TRANSPOSED so every lane owns exactly one softmax
// row (t = laneLo):
//   D2 = A(K) x B(Q)  -> S^T tiles [M=s][N=t]   (16x16x32 f16 WMMA)
//   D3 = A(V) x B(P^T)-> O^T tiles [M=c][N=t]
// P^T B-fragments are built in-register from D2 with one xor-16 ds_swizzle
// exchange (no P LDS round trip). Softmax m/l/alpha are per-lane scalars.
//
// Fast path: one-time f16 prepass of K (-> [b][s][c]) and V (-> [b][c][s])
// into workspace, then K/V chunks staged to LDS with
// global_load_async_to_lds_b128 (ASYNCcnt), double buffered.
// Fallback (workspace too small): register-staged single-buffer variant.
// ---------------------------------------------------------------------------

typedef _Float16 v16h __attribute__((ext_vector_type(16)));
typedef _Float16 h8   __attribute__((ext_vector_type(8)));
typedef float    v8f  __attribute__((ext_vector_type(8)));

#define N_BATCH 64
#define C_DIM   64
#define T_DIM   2048
#define BLOCK_T 128   // 8 waves * 16 rows
#define S_CHUNK 64
#define KP      72    // padded pitch (halfs) for K/V LDS tiles
#define OPITCH  132   // padded pitch (floats) for output staging
#define LOG2E   1.44269504088896340736f
#define SWZ_XOR16 ((16 << 10) | 0x1f)   // group-of-32 xor-16 lane exchange

__device__ __forceinline__ float swz_xor16_f(float x) {
  return __int_as_float(__builtin_amdgcn_ds_swizzle(__float_as_int(x), SWZ_XOR16));
}

// Assemble a v16h A-fragment from two 16B LDS chunks at p and p+gap.
__device__ __forceinline__ v16h ld_frag(const _Float16* p, int gap_halfs) {
  h8 lo = *(const h8*)(p);
  h8 hi = *(const h8*)(p + gap_halfs);
  v16h r;
#pragma unroll
  for (int i = 0; i < 8; ++i) { r[i] = lo[i]; r[i + 8] = hi[i]; }
  return r;
}

// ---------------------------------------------------------------------------
// Per-chunk core: transposed scores, per-lane-scalar online softmax,
// in-register P^T fragment build, transposed PV accumulation.
// ---------------------------------------------------------------------------
__device__ __forceinline__ void attn_chunk(const _Float16* __restrict__ Klds,
                                           const _Float16* __restrict__ Vlds,
                                           const v16h bQ[2], v8f (&accO)[4],
                                           float& m, float& l,
                                           int laneLo, int laneHi) {
  // ---- S^T tiles: D2_j[M = s = j*16 + laneHalfRow][N = t = laneLo] --------
  v8f d2[4];
#pragma unroll
  for (int j = 0; j < 4; ++j) {
    d2[j] = (v8f)0.0f;
#pragma unroll
    for (int g = 0; g < 2; ++g) {
      // A(K): M = s = j*16 + laneLo, K = c = g*32 + laneHi*16 + {0..15}
      const _Float16* ap = &Klds[(j * 16 + laneLo) * KP + g * 32 + laneHi * 16];
      v16h aK = ld_frag(ap, 8);
      d2[j] = __builtin_amdgcn_wmma_f32_16x16x32_f16(false, aK, false, bQ[g],
                                                     (short)0, d2[j], false, false);
    }
  }

  // ---- online softmax: lane owns row t = laneLo; halves hold disjoint s ---
  float x = d2[0][0];
#pragma unroll
  for (int j = 0; j < 4; ++j)
#pragma unroll
    for (int r = 0; r < 8; ++r) x = fmaxf(x, d2[j][r]);
  x = fmaxf(x, swz_xor16_f(x));
  float mnew  = fmaxf(m, x);
  float alpha = __builtin_amdgcn_exp2f(m - mnew);   // exp2 domain; 0 at start
  m = mnew;

  float rsum = 0.0f;
#pragma unroll
  for (int j = 0; j < 4; ++j)
#pragma unroll
    for (int r = 0; r < 8; ++r) {
      float p = __builtin_amdgcn_exp2f(d2[j][r] - mnew);
      d2[j][r] = p;
      rsum += p;
    }
  rsum += swz_xor16_f(rsum);
  l = l * alpha + rsum;
#pragma unroll
  for (int jc = 0; jc < 4; ++jc)
#pragma unroll
    for (int r = 0; r < 8; ++r) accO[jc][r] *= alpha;

  // ---- O^T += A(V) x B(P^T), per 32-wide K(s) step ------------------------
#pragma unroll
  for (int hk = 0; hk < 2; ++hk) {
    // Build B(P^T) fragment: half0 lanes need s = hk*32+0..15 (tile 2hk),
    // half1 lanes need s = hk*32+16..31 (tile 2hk+1). Each lane holds 8 of
    // its 16 values; the other 8 come from lane^16 via one packed swizzle.
    v16h bP;
#pragma unroll
    for (int w = 0; w < 4; ++w) {
      union HU { unsigned u; _Float16 h[2]; };
      HU a0, a1, rc, kp, lo, hi;
      a0.h[0] = (_Float16)d2[2 * hk][2 * w];
      a0.h[1] = (_Float16)d2[2 * hk][2 * w + 1];
      a1.h[0] = (_Float16)d2[2 * hk + 1][2 * w];
      a1.h[1] = (_Float16)d2[2 * hk + 1][2 * w + 1];
      unsigned send = laneHi ? a0.u : a1.u;   // tile the partner half needs
      unsigned keep = laneHi ? a1.u : a0.u;   // tile this half keeps
      rc.u = (unsigned)__builtin_amdgcn_ds_swizzle((int)send, SWZ_XOR16);
      kp.u = keep;
      lo.u = laneHi ? rc.u : kp.u;            // elements 2w,2w+1   (K=2w..)
      hi.u = laneHi ? kp.u : rc.u;            // elements 8+2w,9+2w
      bP[2 * w]     = lo.h[0];
      bP[2 * w + 1] = lo.h[1];
      bP[8 + 2 * w] = hi.h[0];
      bP[9 + 2 * w] = hi.h[1];
    }
#pragma unroll
    for (int jc = 0; jc < 4; ++jc) {
      // A(V): M = c = jc*16 + laneLo, K = s = hk*32 + laneHi*16 + {0..15}
      const _Float16* vp = &Vlds[(jc * 16 + laneLo) * KP + hk * 32 + laneHi * 16];
      v16h aV = ld_frag(vp, 8);
      accO[jc] = __builtin_amdgcn_wmma_f32_16x16x32_f16(false, aV, false, bP,
                                                        (short)0, accO[jc], false, false);
    }
  }
}

// B(Q) fragments: N = t = t0 + laneLo, K = c = g*32 + laneHi*16 + i
__device__ __forceinline__ void load_bQ(const float* __restrict__ q_b,
                                        int t0, int laneLo, int laneHi,
                                        v16h bQ[2]) {
  const float qscale = 0.125f * LOG2E;   // 1/sqrt(C) * log2(e)
#pragma unroll
  for (int g = 0; g < 2; ++g)
#pragma unroll
    for (int i = 0; i < 16; ++i) {
      int c = g * 32 + laneHi * 16 + i;
      bQ[g][i] = (_Float16)(q_b[(size_t)c * T_DIM + t0 + laneLo] * qscale);
    }
}

// Epilogue: normalize O^T tiles, transpose through LDS, coalesced stores.
__device__ __forceinline__ void write_out(float* __restrict__ out, float* Olds,
                                          const v8f (&accO)[4], float l,
                                          int b, int tBlock, int tid,
                                          int wave, int laneLo, int laneHi) {
  float invl = 1.0f / l;
#pragma unroll
  for (int jc = 0; jc < 4; ++jc)
#pragma unroll
    for (int r = 0; r < 8; ++r) {
      int c = jc * 16 + laneHi * 8 + r;
      Olds[c * OPITCH + wave * 16 + laneLo] = accO[jc][r] * invl;
    }
  __syncthreads();
  for (int i = tid; i < 64 * 32; i += 256) {
    int c  = i >> 5;
    int tt = (i & 31) << 2;
    float4 val = *(const float4*)&Olds[c * OPITCH + tt];
    *(float4*)&out[((size_t)(b * C_DIM + c)) * T_DIM + tBlock + tt] = val;
  }
}

// ============================================================================
// Pre-pass A: K plane [c][s] f32 -> Kh [b][s][c] f16  (transpose via LDS)
// ============================================================================
__global__ __launch_bounds__(256)
void prepass_k_transpose(const float* __restrict__ qkv, _Float16* __restrict__ Kh) {
  __shared__ __align__(16) _Float16 T[64 * KP];
  const int b   = blockIdx.y;
  const int s0  = blockIdx.x * 64;
  const int tid = threadIdx.x;
  const float* k_b = qkv + (size_t)b * 3 * C_DIM * T_DIM + (size_t)C_DIM * T_DIM;

  const int c0 = (tid >> 3) << 1;
  const int s8 = (tid & 7) << 3;
  const float* kp0 = k_b + (size_t)c0 * T_DIM + s0 + s8;
  const float* kp1 = kp0 + T_DIM;
  float ka[8], kb[8];
  *(float4*)&ka[0] = *(const float4*)(kp0);
  *(float4*)&ka[4] = *(const float4*)(kp0 + 4);
  *(float4*)&kb[0] = *(const float4*)(kp1);
  *(float4*)&kb[4] = *(const float4*)(kp1 + 4);
#pragma unroll
  for (int i = 0; i < 8; ++i) {
    union { _Float16 h[2]; unsigned int u; } pk;
    pk.h[0] = (_Float16)ka[i];
    pk.h[1] = (_Float16)kb[i];
    *(unsigned int*)&T[(s8 + i) * KP + c0] = pk.u;
  }
  __syncthreads();

  const int s    = tid >> 2;
  const int cseg = (tid & 3) << 4;
  h8 lo = *(const h8*)&T[s * KP + cseg];
  h8 hi = *(const h8*)&T[s * KP + cseg + 8];
  _Float16* dst = Kh + ((size_t)b * T_DIM + s0 + s) * C_DIM + cseg;
  *(h8*)(dst)     = lo;
  *(h8*)(dst + 8) = hi;
}

// ============================================================================
// Pre-pass B: V plane [c][s] f32 -> Vh [b][c][s] f16
// ============================================================================
__global__ __launch_bounds__(256)
void prepass_v_convert(const float* __restrict__ qkv, _Float16* __restrict__ Vh) {
  const size_t gid = (size_t)blockIdx.x * 256 + threadIdx.x;
  const size_t e   = gid * 8;
  const size_t b   = e >> 17;            // 131072 elems per plane
  const size_t rem = e & 131071;
  const float* src = qkv + b * (3 * C_DIM * T_DIM) + (size_t)2 * C_DIM * T_DIM + rem;
  float va[8];
  *(float4*)&va[0] = *(const float4*)(src);
  *(float4*)&va[4] = *(const float4*)(src + 4);
  h8 h;
#pragma unroll
  for (int i = 0; i < 8; ++i) h[i] = (_Float16)va[i];
  *(h8*)(Vh + b * 131072 + rem) = h;
}

// ============================================================================
// Main kernel (async staged, double buffered)
// LDS (bytes): Kbuf0 @0, Kbuf1 @9216, Vbuf0 @18432, Vbuf1 @27648. Total 36864.
// Epilogue overlays Olds [64][OPITCH] f32 @0 (33792 B).
// ============================================================================
__device__ __forceinline__ void stage_async(const _Float16* __restrict__ Kh_b,
                                            const _Float16* __restrict__ Vh_b,
                                            unsigned kbase, unsigned vbase,
                                            int s0, int tid) {
#pragma unroll
  for (int j = 0; j < 2; ++j) {
    int id  = tid + j * 256;
    int row = id >> 3;                 // 0..63 (s for K, c for V)
    int seg = (id & 7) << 3;           // 8-half (16B) segment
    unsigned koff = kbase + (unsigned)((row * KP + seg) * 2);
    const _Float16* kg = Kh_b + (size_t)(s0 + row) * C_DIM + seg;
    asm volatile("global_load_async_to_lds_b128 %0, %1, off"
                 :: "v"(koff), "v"(kg) : "memory");
    unsigned voff = vbase + (unsigned)((row * KP + seg) * 2);
    const _Float16* vg = Vh_b + (size_t)row * T_DIM + s0 + seg;
    asm volatile("global_load_async_to_lds_b128 %0, %1, off"
                 :: "v"(voff), "v"(vg) : "memory");
  }
}

__global__ __launch_bounds__(256)
void qkv_attention_async(const float* __restrict__ qkv,
                         const _Float16* __restrict__ Kh,
                         const _Float16* __restrict__ Vh,
                         float* __restrict__ out) {
  __shared__ __align__(16) unsigned char smem[36864];

  const int b      = blockIdx.y;
  const int tBlock = blockIdx.x * BLOCK_T;
  const int tid    = threadIdx.x;
  const int wave   = tid >> 5;
  const int lane   = tid & 31;
  const int laneLo = lane & 15;
  const int laneHi = lane >> 4;
  const int t0     = tBlock + wave * 16;

  const float*    q_b  = qkv + (size_t)b * 3 * C_DIM * T_DIM;
  const _Float16* Kh_b = Kh + (size_t)b * T_DIM * C_DIM;
  const _Float16* Vh_b = Vh + (size_t)b * C_DIM * T_DIM;
  const unsigned  lds0 = (unsigned)(unsigned long long)(void*)smem;

  v16h bQ[2];
  load_bQ(q_b, t0, laneLo, laneHi, bQ);

  v8f accO[4];
#pragma unroll
  for (int jc = 0; jc < 4; ++jc) accO[jc] = (v8f)0.0f;
  float m = -INFINITY, l = 0.0f;

  stage_async(Kh_b, Vh_b, lds0 + 0, lds0 + 18432, 0, tid);
  asm volatile("s_wait_asynccnt 0" ::: "memory");
  __syncthreads();

  for (int ci = 0; ci < T_DIM / S_CHUNK; ++ci) {
    const int cur = ci & 1;
    if (ci + 1 < T_DIM / S_CHUNK) {
      stage_async(Kh_b, Vh_b,
                  lds0 + (cur ^ 1) * 9216,
                  lds0 + 18432 + (cur ^ 1) * 9216,
                  (ci + 1) * S_CHUNK, tid);
    }
    const _Float16* Klds = (const _Float16*)(smem + cur * 9216);
    const _Float16* Vlds = (const _Float16*)(smem + 18432 + cur * 9216);

    attn_chunk(Klds, Vlds, bQ, accO, m, l, laneLo, laneHi);

    asm volatile("s_wait_asynccnt 0" ::: "memory");
    __syncthreads();
  }

  write_out(out, (float*)smem, accO, l, b, tBlock, tid, wave, laneLo, laneHi);
}

// ============================================================================
// Fallback kernel (no workspace): register-staged, single buffered
// ============================================================================
__global__ __launch_bounds__(256)
void qkv_attention_fallback(const float* __restrict__ qkv, float* __restrict__ out) {
  __shared__ __align__(16) unsigned char smem[33792];   // Olds overlay size
  _Float16* Klds = (_Float16*)(smem);
  _Float16* Vlds = (_Float16*)(smem + 64 * KP * 2);

  const int b      = blockIdx.y;
  const int tBlock = blockIdx.x * BLOCK_T;
  const int tid    = threadIdx.x;
  const int wave   = tid >> 5;
  const int lane   = tid & 31;
  const int laneLo = lane & 15;
  const int laneHi = lane >> 4;
  const int t0     = tBlock + wave * 16;

  const float* q_b = qkv + (size_t)b * 3 * C_DIM * T_DIM;
  const float* k_b = q_b + (size_t)C_DIM * T_DIM;
  const float* v_b = q_b + (size_t)2 * C_DIM * T_DIM;

  v16h bQ[2];
  load_bQ(q_b, t0, laneLo, laneHi, bQ);

  v8f accO[4];
#pragma unroll
  for (int jc = 0; jc < 4; ++jc) accO[jc] = (v8f)0.0f;
  float m = -INFINITY, l = 0.0f;

  const int kc0  = (tid >> 3) << 1;
  const int ks8  = (tid & 7) << 3;
  const int vc   = tid >> 2;
  const int vs16 = (tid & 3) << 4;

  for (int s0 = 0; s0 < T_DIM; s0 += S_CHUNK) {
    {
      const float* kp0 = k_b + (size_t)kc0 * T_DIM + s0 + ks8;
      const float* kp1 = kp0 + T_DIM;
      float ka[8], kb2[8];
      *(float4*)&ka[0]  = *(const float4*)(kp0);
      *(float4*)&ka[4]  = *(const float4*)(kp0 + 4);
      *(float4*)&kb2[0] = *(const float4*)(kp1);
      *(float4*)&kb2[4] = *(const float4*)(kp1 + 4);
#pragma unroll
      for (int i = 0; i < 8; ++i) {
        union { _Float16 h[2]; unsigned int u; } pk;
        pk.h[0] = (_Float16)ka[i];
        pk.h[1] = (_Float16)kb2[i];
        *(unsigned int*)&Klds[(ks8 + i) * KP + kc0] = pk.u;
      }
    }
    {
      const float* vp = v_b + (size_t)vc * T_DIM + s0 + vs16;
      float va[16];
      *(float4*)&va[0]  = *(const float4*)(vp);
      *(float4*)&va[4]  = *(const float4*)(vp + 4);
      *(float4*)&va[8]  = *(const float4*)(vp + 8);
      *(float4*)&va[12] = *(const float4*)(vp + 12);
      h8 h0, h1;
#pragma unroll
      for (int i = 0; i < 8; ++i) { h0[i] = (_Float16)va[i]; h1[i] = (_Float16)va[8 + i]; }
      *(h8*)&Vlds[vc * KP + vs16]     = h0;
      *(h8*)&Vlds[vc * KP + vs16 + 8] = h1;
    }
    __syncthreads();

    attn_chunk(Klds, Vlds, bQ, accO, m, l, laneLo, laneHi);

    __syncthreads();
  }

  write_out(out, (float*)smem, accO, l, b, tBlock, tid, wave, laneLo, laneHi);
}

// ============================================================================
extern "C" void kernel_launch(void* const* d_in, const int* in_sizes, int n_in,
                              void* d_out, int out_size, void* d_ws, size_t ws_size,
                              hipStream_t stream) {
  (void)in_sizes; (void)n_in; (void)out_size;
  const float* qkv = (const float*)d_in[0];
  float* out = (float*)d_out;

  const size_t plane = (size_t)N_BATCH * T_DIM * C_DIM;      // elements
  const size_t need  = 2 * plane * sizeof(_Float16);         // Kh + Vh, 33.5 MB

  if (ws_size >= need && d_ws != nullptr) {
    _Float16* Kh = (_Float16*)d_ws;
    _Float16* Vh = Kh + plane;
    prepass_k_transpose<<<dim3(T_DIM / 64, N_BATCH), 256, 0, stream>>>(qkv, Kh);
    prepass_v_convert<<<dim3(4096), 256, 0, stream>>>(qkv, Vh);
    qkv_attention_async<<<dim3(T_DIM / BLOCK_T, N_BATCH), 256, 0, stream>>>(qkv, Kh, Vh, out);
  } else {
    qkv_attention_fallback<<<dim3(T_DIM / BLOCK_T, N_BATCH), 256, 0, stream>>>(qkv, out);
  }
}
